// CoTModule_51977694216412
// MI455X (gfx1250) — compile-verified, matrix operand order checked by reading, not run on hardware
//
#include <hip/hip_runtime.h>

// ---------------------------------------------------------------------------
// CoT transformer for MI455X (gfx1250, wave32, WMMA).
//  - GEMMs: v_wmma_f32_16x16x32_f16, one wave computes a 16x64 strip
//    (4 accumulators reuse one A fragment -> 10 b128 loads per 4 WMMAs).
//  - Weights converted once per call to transposed [N x K] f16; wq/wk/wv are
//    packed contiguously so QKV is a single N=1536 GEMM (cross wk/wv: N=1024).
//  - global_prefetch_b8 hints for the next K-chunk of A/B streams.
//  - Decode uses exact incremental KV-cache decoding (causal mask => exact).
// ---------------------------------------------------------------------------

typedef _Float16 half_t;
typedef __attribute__((ext_vector_type(8)))  half_t v8h;
typedef __attribute__((ext_vector_type(16))) half_t v16h;
typedef __attribute__((ext_vector_type(8)))  float  v8f;

constexpr int kB   = 32;
constexpr int kTin = 128;
constexpr int kD   = 512;
constexpr int kH   = 8;
constexpr int kHD  = 64;
constexpr int kFF  = 2048;
constexpr int kLin = 4;
constexpr int kLcot= 4;
constexpr int kTC  = 32;
constexpr int kVC  = 1024;
constexpr int kTD  = kTC + 1;      // 33 decoder positions
constexpr int kM1  = kB * kTin;    // 4096 encoder rows
constexpr int kWaves = 4;          // waves per GEMM block (each wave: 16x64)

// ---------------------------- small utility kernels ------------------------

__global__ void k_cvt_w_t(const float* __restrict__ W, half_t* __restrict__ Wt,
                          int K, int N) {
  int idx = blockIdx.x * 256 + threadIdx.x;      // over K*N outputs
  if (idx >= K * N) return;
  int n = idx / K, k = idx - n * K;
  Wt[idx] = (half_t)W[(size_t)k * N + n];        // Wt[n*K+k] = W[k*N+n]
}

__global__ void k_cvt16(const float* __restrict__ x, half_t* __restrict__ y, int n) {
  int idx = blockIdx.x * 256 + threadIdx.x;
  if (idx < n) y[idx] = (half_t)x[idx];
}

__global__ void k_embed_enc(const int* __restrict__ toks,
                            const float* __restrict__ tok_emb,
                            const float* __restrict__ pos_emb,
                            float* __restrict__ x) {
  int idx = blockIdx.x * 256 + threadIdx.x;      // over kM1*kD
  int row = idx / kD, d = idx - row * kD;
  int t = row % kTin;
  x[idx] = tok_emb[(size_t)toks[row] * kD + d] + pos_emb[(size_t)t * kD + d];
}

__global__ void k_embed_dec(const int* __restrict__ cot,
                            const float* __restrict__ tok_emb,
                            const float* __restrict__ pos_emb,
                            float* __restrict__ x, int i) {
  int idx = blockIdx.x * 256 + threadIdx.x;      // over kB*kD
  int b = idx / kD, d = idx - b * kD;
  x[idx] = tok_emb[(size_t)cot[b * kTD + i] * kD + d] + pos_emb[(size_t)i * kD + d];
}

__global__ void k_init_cot(int* __restrict__ cot) {
  int idx = blockIdx.x * 256 + threadIdx.x;
  if (idx < kB * kTD) cot[idx] = (idx % kTD == 0) ? kVC : 0;
}

// qkv2 is a packed [B x 3D] buffer: q at +0, k at +D, v at +2D
__global__ void k_append_kv(const float* __restrict__ qkv2,
                            float* __restrict__ kc, float* __restrict__ vc, int i) {
  int idx = blockIdx.x * 256 + threadIdx.x;      // over kB*kD
  int b = idx / kD, d = idx - b * kD;
  size_t off = ((size_t)b * kTD + i) * kD + d;
  kc[off] = qkv2[(size_t)b * (3 * kD) + kD + d];
  vc[off] = qkv2[(size_t)b * (3 * kD) + 2 * kD + d];
}

// ------------------------------- LayerNorm ---------------------------------
// One wave per row (D=512 -> 16 values/lane), f16 output feeding WMMA A.
__global__ void k_ln(const float* __restrict__ x, const float* __restrict__ g,
                     half_t* __restrict__ y, int M) {
  int row = blockIdx.x * blockDim.y + threadIdx.y;
  if (row >= M) return;
  int lane = threadIdx.x;
  const float* xr = x + (size_t)row * kD;
  float v[16];
  float s = 0.f;
  #pragma unroll
  for (int i = 0; i < 16; ++i) { v[i] = xr[lane + 32 * i]; s += v[i]; }
  #pragma unroll
  for (int o = 16; o; o >>= 1) s += __shfl_xor(s, o, 32);
  float mean = s * (1.f / kD);
  float var = 0.f;
  #pragma unroll
  for (int i = 0; i < 16; ++i) { float d = v[i] - mean; var += d * d; }
  #pragma unroll
  for (int o = 16; o; o >>= 1) var += __shfl_xor(var, o, 32);
  float rstd = rsqrtf(var * (1.f / kD) + 1e-6f);
  half_t* yr = y + (size_t)row * kD;
  #pragma unroll
  for (int i = 0; i < 16; ++i) {
    int c = lane + 32 * i;
    yr[c] = (half_t)((v[i] - mean) * rstd * g[c]);
  }
}

// ------------------------------- WMMA GEMM ---------------------------------
// C[M,N] = A[M,K] * Bt[N,K]^T. One wave = 16x64 strip: one A fragment feeds
// four 16x16 accumulators. Fragment loads follow the CDNA5 16-bit layout:
// lane (l&15) holds the row/col, lanes>=16 carry the K+8 / K+24 halves, so
// each fragment is two contiguous b128 loads per lane.
__device__ __forceinline__ void wmma_strip4(const half_t* __restrict__ Ar,
                                            const half_t* __restrict__ Br0,
                                            int Kd, v8f acc[4]) {
  const int hi = threadIdx.x >> 4;
  const size_t tstride = (size_t)16 * Kd;        // 16 rows of Bt per subtile
  #pragma unroll 2
  for (int k0 = 0; k0 < Kd; k0 += 32) {
    v8h a0 = *(const v8h*)(Ar + k0 + 8 * hi);
    v8h a1 = *(const v8h*)(Ar + k0 + 16 + 8 * hi);
    v16h av = __builtin_shufflevector(a0, a1, 0,1,2,3,4,5,6,7,8,9,10,11,12,13,14,15);
    __builtin_prefetch((const void*)(Ar  + k0 + 256), 0, 1);  // global_prefetch_b8
    __builtin_prefetch((const void*)(Br0 + k0 + 256), 0, 1);
    #pragma unroll
    for (int j = 0; j < 4; ++j) {
      const half_t* Br = Br0 + (size_t)j * tstride;
      v8h b0 = *(const v8h*)(Br + k0 + 8 * hi);
      v8h b1 = *(const v8h*)(Br + k0 + 16 + 8 * hi);
      v16h bv = __builtin_shufflevector(b0, b1, 0,1,2,3,4,5,6,7,8,9,10,11,12,13,14,15);
      acc[j] = __builtin_amdgcn_wmma_f32_16x16x32_f16(false, av, false, bv,
                                                      (short)0, acc[j], false, false);
    }
  }
}

__global__ void k_gemm(const half_t* __restrict__ A, const half_t* __restrict__ Bt,
                       float* __restrict__ C, const float* __restrict__ resid,
                       int M, int N, int Kd) {
  int lane   = threadIdx.x;
  int tileN0 = (blockIdx.x * kWaves + threadIdx.y) * 4;   // 16-col subtiles
  int tileM  = blockIdx.y;
  if (tileN0 * 16 >= N) return;                           // wave-uniform
  const half_t* Ar  = A  + (size_t)(tileM * 16 + (lane & 15)) * Kd;
  const half_t* Br0 = Bt + (size_t)(tileN0 * 16 + (lane & 15)) * Kd;
  v8f acc[4] = {};
  wmma_strip4(Ar, Br0, Kd, acc);
  // D layout: VGPR r -> row r (lanes 0-15) / row r+8 (lanes 16-31); col = lane&15
  int rb = tileM * 16 + ((lane >> 4) ? 8 : 0);
  #pragma unroll
  for (int j = 0; j < 4; ++j) {
    int cn = (tileN0 + j) * 16 + (lane & 15);
    #pragma unroll
    for (int r = 0; r < 8; ++r) {
      size_t off = (size_t)(rb + r) * N + cn;
      float v = acc[j][r];
      if (resid) v += resid[off];
      C[off] = v;
    }
  }
}

__device__ __forceinline__ float gelu_tanh(float x) {
  return 0.5f * x * (1.f + tanhf(0.7978845608028654f * (x + 0.044715f * x * x * x)));
}

__global__ void k_gemm_gelu16(const half_t* __restrict__ A, const half_t* __restrict__ Bt,
                              half_t* __restrict__ C, int M, int N, int Kd) {
  int lane   = threadIdx.x;
  int tileN0 = (blockIdx.x * kWaves + threadIdx.y) * 4;
  int tileM  = blockIdx.y;
  if (tileN0 * 16 >= N) return;
  const half_t* Ar  = A  + (size_t)(tileM * 16 + (lane & 15)) * Kd;
  const half_t* Br0 = Bt + (size_t)(tileN0 * 16 + (lane & 15)) * Kd;
  v8f acc[4] = {};
  wmma_strip4(Ar, Br0, Kd, acc);
  int rb = tileM * 16 + ((lane >> 4) ? 8 : 0);
  #pragma unroll
  for (int j = 0; j < 4; ++j) {
    int cn = (tileN0 + j) * 16 + (lane & 15);
    #pragma unroll
    for (int r = 0; r < 8; ++r)
      C[(size_t)(rb + r) * N + cn] = (half_t)gelu_tanh(acc[j][r]);
  }
}

// ------------------------------- attention ---------------------------------
// One wave per (b, query-position, head). Scores staged in LDS, softmax via
// wave32 shuffles, f16 output rows (pitch kD) feed the projection WMMA GEMM.
// qPitch/kvPitch allow reading directly from packed QKV / packed cross-KV.
__global__ void k_attn(const float* __restrict__ Q, const float* __restrict__ K,
                       const float* __restrict__ V, half_t* __restrict__ O,
                       int Tq, int Tk, int kvStride, int qPitch, int kvPitch) {
  int idx = blockIdx.x;
  int h = idx % kH;
  int t = (idx / kH) % Tq;
  int b = idx / (kH * Tq);
  int lane = threadIdx.x;
  __shared__ float qs[kHD];
  __shared__ float sc[kTin];
  const float* qp = Q + (size_t)(b * Tq + t) * qPitch + h * kHD;
  qs[lane]      = qp[lane];
  qs[lane + 32] = qp[lane + 32];
  __syncthreads();
  float mx = -1e30f;
  for (int k = lane; k < Tk; k += 32) {
    const float* kp = K + (size_t)(b * kvStride + k) * kvPitch + h * kHD;
    float s = 0.f;
    #pragma unroll
    for (int d = 0; d < kHD; ++d) s += qs[d] * kp[d];
    s *= 0.125f;                                  // 1/sqrt(64)
    sc[k] = s;
    mx = fmaxf(mx, s);
  }
  #pragma unroll
  for (int o = 16; o; o >>= 1) mx = fmaxf(mx, __shfl_xor(mx, o, 32));
  __syncthreads();
  float sum = 0.f;
  for (int k = lane; k < Tk; k += 32) {
    float e = __expf(sc[k] - mx);
    sc[k] = e;
    sum += e;
  }
  #pragma unroll
  for (int o = 16; o; o >>= 1) sum += __shfl_xor(sum, o, 32);
  float inv = 1.f / sum;
  __syncthreads();
  float a0 = 0.f, a1 = 0.f;
  for (int k = 0; k < Tk; ++k) {
    const float2 vv = *(const float2*)(V + (size_t)(b * kvStride + k) * kvPitch
                                         + h * kHD + lane * 2);
    float w = sc[k];
    a0 += w * vv.x;
    a1 += w * vv.y;
  }
  half_t* op = O + (size_t)(b * Tq + t) * kD + h * kHD + lane * 2;
  op[0] = (half_t)(a0 * inv);
  op[1] = (half_t)(a1 * inv);
}

// ------------------------- argmax + output emission -------------------------
__global__ void k_argmax_emit(const float* __restrict__ logits, int* __restrict__ cot,
                              int* __restrict__ out_tok, float* __restrict__ out_logits,
                              int i) {
  int b = blockIdx.x, lane = threadIdx.x;
  const float* lr = logits + (size_t)b * kVC;
  float best = -3.4e38f; int bi = 0;
  for (int v = lane; v < kVC; v += 32) {
    float val = lr[v];
    if (val > best) { best = val; bi = v; }      // '>' keeps first max
  }
  #pragma unroll
  for (int o = 16; o; o >>= 1) {
    float ov = __shfl_xor(best, o, 32);
    int   oi = __shfl_xor(bi,   o, 32);
    if (ov > best || (ov == best && oi < bi)) { best = ov; bi = oi; }
  }
  if (lane == 0) {
    cot[b * kTD + i + 1] = bi;
    out_tok[b * kTC + i] = bi;
  }
  float* dst = out_logits + ((size_t)b * kTC + i) * kVC;
  for (int v = lane; v < kVC; v += 32) dst[v] = lr[v];
}

// ------------------------------- host driver --------------------------------

extern "C" void kernel_launch(void* const* d_in, const int* in_sizes, int n_in,
                              void* d_out, int out_size, void* d_ws, size_t ws_size,
                              hipStream_t stream) {
  (void)in_sizes; (void)n_in; (void)out_size; (void)ws_size;
  auto F = [&](int i) { return (const float*)d_in[i]; };

  // ---- input leaves, JAX sorted-key pytree order (91 leaves) ----
  const int* inputs = (const int*)d_in[0];
  struct DecW { const float *a_wk,*a_wo,*a_wq,*a_wv,
                            *c_wk,*c_wo,*c_wq,*c_wv,
                            *ln1,*lnc,*lnm,*w1,*w2; };
  struct EncW { const float *a_wk,*a_wo,*a_wq,*a_wv,*ln1,*lnm,*w1,*w2; };
  DecW dec[kLcot]; EncW enc[kLin];
  for (int l = 0; l < kLcot; ++l) {
    int s = 1 + 13 * l;
    dec[l] = { F(s+0),F(s+1),F(s+2),F(s+3), F(s+4),F(s+5),F(s+6),F(s+7),
               F(s+8),F(s+9),F(s+10),F(s+11),F(s+12) };
  }
  const float* headW = F(53);
  for (int l = 0; l < kLin; ++l) {
    int s = 54 + 8 * l;
    enc[l] = { F(s+0),F(s+1),F(s+2),F(s+3),F(s+4),F(s+5),F(s+6),F(s+7) };
  }
  const float* ln_f    = F(86);
  const float* pos_cot = F(87);
  const float* pos_in  = F(88);
  const float* tok_cot = F(89);
  const float* tok_in  = F(90);

  // ---- workspace bump allocator (same layout every call: deterministic) ----
  char* base = (char*)d_ws; size_t off = 0;
  auto alloc = [&](size_t bytes) -> void* {
    void* r = base + off; off = (off + bytes + 255) & ~(size_t)255; return r;
  };
  const size_t DD = (size_t)kD * kD, DF = (size_t)kD * kFF, DV = (size_t)kD * kVC;

  // f16 transposed weights; wq/wk/wv packed (N=1536), cross wk/wv packed (N=1024)
  half_t *encQKV[kLin], *encWo[kLin], *encW1[kLin], *encW2[kLin];
  for (int l = 0; l < kLin; ++l) {
    encQKV[l] = (half_t*)alloc(3 * DD * 2);
    encWo[l]  = (half_t*)alloc(DD * 2);
    encW1[l]  = (half_t*)alloc(DF * 2);
    encW2[l]  = (half_t*)alloc(DF * 2);
  }
  half_t *decQKV[kLcot], *decWo[kLcot], *decCq[kLcot], *decCKV[kLcot],
         *decCo[kLcot], *decW1[kLcot], *decW2[kLcot];
  for (int l = 0; l < kLcot; ++l) {
    decQKV[l] = (half_t*)alloc(3 * DD * 2);
    decWo[l]  = (half_t*)alloc(DD * 2);
    decCq[l]  = (half_t*)alloc(DD * 2);
    decCKV[l] = (half_t*)alloc(2 * DD * 2);
    decCo[l]  = (half_t*)alloc(DD * 2);
    decW1[l]  = (half_t*)alloc(DF * 2);
    decW2[l]  = (half_t*)alloc(DF * 2);
  }
  half_t* headT = (half_t*)alloc(DV * 2);

  float*  x     = (float*) alloc((size_t)kM1 * kD * 4);
  half_t* enc16 = (half_t*)alloc((size_t)kM1 * kD * 2);
  half_t* ln16  = (half_t*)alloc((size_t)kM1 * kD * 2);
  float*  qkv   = (float*) alloc((size_t)kM1 * 3 * kD * 4);   // packed q|k|v
  half_t* att16 = (half_t*)alloc((size_t)kM1 * kD * 2);
  half_t* h16   = (half_t*)alloc((size_t)kM1 * kFF * 2);
  float *ckv[kLcot], *kcache[kLcot], *vcache[kLcot];
  for (int l = 0; l < kLcot; ++l)
    ckv[l] = (float*)alloc((size_t)kM1 * 2 * kD * 4);         // packed Kc|Vc
  for (int l = 0; l < kLcot; ++l) {
    kcache[l] = (float*)alloc((size_t)kB * kTD * kD * 4);
    vcache[l] = (float*)alloc((size_t)kB * kTD * kD * 4);
  }
  float*  x2    = (float*) alloc((size_t)kB * kD * 4);
  half_t* x2ln  = (half_t*)alloc((size_t)kB * kD * 2);
  float*  qkv2  = (float*) alloc((size_t)kB * 3 * kD * 4);
  float*  q2    = (float*) alloc((size_t)kB * kD * 4);
  half_t* a216  = (half_t*)alloc((size_t)kB * kD * 2);
  half_t* h216  = (half_t*)alloc((size_t)kB * kFF * 2);
  float*  logits= (float*) alloc((size_t)kB * kVC * 4);
  int*    cot   = (int*)   alloc((size_t)kB * kTD * 4);

  // ---- launch helpers ----
  auto cvt = [&](const float* W, half_t* Wt, int K, int N) {
    int n = K * N;
    k_cvt_w_t<<<(n + 255) / 256, 256, 0, stream>>>(W, Wt, K, N);
  };
  auto gemm = [&](const half_t* A, const half_t* Bt, float* C, const float* resid,
                  int M, int N, int Kd) {
    dim3 blk(32, kWaves);
    dim3 grd((N / 64 + kWaves - 1) / kWaves, M / 16);
    k_gemm<<<grd, blk, 0, stream>>>(A, Bt, C, resid, M, N, Kd);
  };
  auto gemm_gelu = [&](const half_t* A, const half_t* Bt, half_t* C,
                       int M, int N, int Kd) {
    dim3 blk(32, kWaves);
    dim3 grd((N / 64 + kWaves - 1) / kWaves, M / 16);
    k_gemm_gelu16<<<grd, blk, 0, stream>>>(A, Bt, C, M, N, Kd);
  };
  auto lnorm = [&](const float* xp, const float* g, half_t* y, int M) {
    dim3 blk(32, 8);
    k_ln<<<(M + 7) / 8, blk, 0, stream>>>(xp, g, y, M);
  };

  // ---- weight conversion (f32 -> transposed f16, ~57 MB, L2-resident) ----
  for (int l = 0; l < kLin; ++l) {
    cvt(enc[l].a_wq, encQKV[l] + 0 * DD, kD, kD);
    cvt(enc[l].a_wk, encQKV[l] + 1 * DD, kD, kD);
    cvt(enc[l].a_wv, encQKV[l] + 2 * DD, kD, kD);
    cvt(enc[l].a_wo, encWo[l], kD, kD);
    cvt(enc[l].w1,   encW1[l], kD, kFF);
    cvt(enc[l].w2,   encW2[l], kFF, kD);
  }
  for (int l = 0; l < kLcot; ++l) {
    cvt(dec[l].a_wq, decQKV[l] + 0 * DD, kD, kD);
    cvt(dec[l].a_wk, decQKV[l] + 1 * DD, kD, kD);
    cvt(dec[l].a_wv, decQKV[l] + 2 * DD, kD, kD);
    cvt(dec[l].a_wo, decWo[l], kD, kD);
    cvt(dec[l].c_wq, decCq[l], kD, kD);
    cvt(dec[l].c_wk, decCKV[l] + 0 * DD, kD, kD);
    cvt(dec[l].c_wv, decCKV[l] + 1 * DD, kD, kD);
    cvt(dec[l].c_wo, decCo[l], kD, kD);
    cvt(dec[l].w1,   decW1[l], kD, kFF);
    cvt(dec[l].w2,   decW2[l], kFF, kD);
  }
  cvt(headW, headT, kD, kVC);

  // ---- encoder ----
  k_embed_enc<<<(kM1 * kD) / 256, 256, 0, stream>>>(inputs, tok_in, pos_in, x);
  for (int l = 0; l < kLin; ++l) {
    lnorm(x, enc[l].ln1, ln16, kM1);
    gemm(ln16, encQKV[l], qkv, nullptr, kM1, 3 * kD, kD);
    k_attn<<<kB * kTin * kH, 32, 0, stream>>>(qkv, qkv + kD, qkv + 2 * kD, att16,
                                              kTin, kTin, kTin, 3 * kD, 3 * kD);
    gemm(att16, encWo[l], x, x, kM1, kD, kD);
    lnorm(x, enc[l].lnm, ln16, kM1);
    gemm_gelu(ln16, encW1[l], h16, kM1, kFF, kD);
    gemm(h16, encW2[l], x, x, kM1, kD, kFF);
  }
  k_cvt16<<<(kM1 * kD) / 256, 256, 0, stream>>>(x, enc16, kM1 * kD);

  // cross-attention K/V depend only on enc: compute once per layer (packed)
  for (int l = 0; l < kLcot; ++l)
    gemm(enc16, decCKV[l], ckv[l], nullptr, kM1, 2 * kD, kD);
  k_init_cot<<<(kB * kTD + 255) / 256, 256, 0, stream>>>(cot);

  int*   out_tok    = (int*)d_out;                       // cot[:,1:] (int32)
  float* out_logits = (float*)d_out + (size_t)kB * kTC;  // logits (f32)

  // ---- incremental decode (exact under the causal mask) ----
  for (int i = 0; i < kTC; ++i) {
    k_embed_dec<<<(kB * kD) / 256, 256, 0, stream>>>(cot, tok_cot, pos_cot, x2, i);
    for (int l = 0; l < kLcot; ++l) {
      lnorm(x2, dec[l].ln1, x2ln, kB);
      gemm(x2ln, decQKV[l], qkv2, nullptr, kB, 3 * kD, kD);
      k_append_kv<<<(kB * kD) / 256, 256, 0, stream>>>(qkv2, kcache[l], vcache[l], i);
      k_attn<<<kB * kH, 32, 0, stream>>>(qkv2, kcache[l], vcache[l], a216,
                                         1, i + 1, kTD, 3 * kD, kD);
      gemm(a216, decWo[l], x2, x2, kB, kD, kD);
      lnorm(x2, dec[l].lnc, x2ln, kB);
      gemm(x2ln, decCq[l], q2, nullptr, kB, kD, kD);
      k_attn<<<kB * kH, 32, 0, stream>>>(q2, ckv[l], ckv[l] + kD, a216,
                                         1, kTin, kTin, kD, 2 * kD);
      gemm(a216, decCo[l], x2, x2, kB, kD, kD);
      lnorm(x2, dec[l].lnm, x2ln, kB);
      gemm_gelu(x2ln, decW1[l], h216, kB, kFF, kD);
      gemm(h216, decW2[l], x2, x2, kB, kD, kFF);
    }
    lnorm(x2, ln_f, x2ln, kB);
    gemm(x2ln, headT, logits, nullptr, kB, kVC, kD);
    k_argmax_emit<<<kB, 32, 0, stream>>>(logits, cot, out_tok, out_logits, i);
  }
}
